// DLRM_Net_41721312314377
// MI455X (gfx1250) — matrix operand order repaired and checked by reading, not compiled, not run on hardware
//
#include <hip/hip_runtime.h>
#include <hip/hip_bf16.h>

// ---------------------------------------------------------------------------
// DLRM forward for MI455X (gfx1250, wave32, WMMA).
// GEMMs: LDS-staged, double-buffered (async global->LDS copies when available),
// 128x128 block tile, 64x32 wave tile, v_wmma_f32_16x16x32_f16.
// ---------------------------------------------------------------------------

typedef __attribute__((ext_vector_type(16))) _Float16 v16h;
typedef __attribute__((ext_vector_type(8)))  _Float16 v8h;
typedef __attribute__((ext_vector_type(8)))  float    v8f;
typedef __attribute__((__vector_size__(16))) int      v4i;

union V16u { v16h v; v8h h[2]; };

__device__ __forceinline__ v16h frag_from(const _Float16* p0, const _Float16* p1) {
  V16u u;
  u.h[0] = *(const v8h*)p0;
  u.h[1] = *(const v8h*)p1;
  return u.v;
}

__device__ __forceinline__ v8f wmma16x16x32(v16h a, v16h b, v8f c) {
  return __builtin_amdgcn_wmma_f32_16x16x32_f16(false, a, false, b, (short)0, c,
                                                false, false);
}

// ---------------------------------------------------------------------------
// Async global -> LDS copy (16 bytes), CDNA5 ASYNCcnt path with fallback.
// Builtin param 0 is 'int4 __device__*' per clang's diagnostic, so cast to
// address_space(1)/(3) pointers-to-int4.
// ---------------------------------------------------------------------------
#if defined(__has_builtin)
#if __has_builtin(__builtin_amdgcn_global_load_async_to_lds_b128)
#define HAVE_ASYNC_LDS 1
#endif
#endif
#ifndef HAVE_ASYNC_LDS
#define HAVE_ASYNC_LDS 0
#endif

__device__ __forceinline__ void copy16_g2l(const _Float16* g, _Float16* l) {
#if HAVE_ASYNC_LDS
  __builtin_amdgcn_global_load_async_to_lds_b128(
      (__attribute__((address_space(1))) v4i*)g,
      (__attribute__((address_space(3))) v4i*)l, 0, 0);
#else
  *(v8h*)l = *(const v8h*)g;
#endif
}

__device__ __forceinline__ void wait_async_copies() {
#if HAVE_ASYNC_LDS
#if __has_builtin(__builtin_amdgcn_s_wait_asynccnt)
  __builtin_amdgcn_s_wait_asynccnt(0);
#else
  asm volatile("s_wait_asynccnt 0" ::: "memory");
#endif
#endif
}

// ---------------------------------------------------------------------------
// f32 -> f16 convert with K-padding (pads columns [kin, kp) with zero)
// ---------------------------------------------------------------------------
__global__ __launch_bounds__(256) void cvt_pad(const float* __restrict__ src,
                                               _Float16* __restrict__ dst,
                                               int rows, int kin, int kp) {
  int i = blockIdx.x * 256 + threadIdx.x;
  if (i >= rows * kp) return;
  int r = i / kp, k = i - r * kp;
  dst[i] = (k < kin) ? (_Float16)src[(size_t)r * kin + k] : (_Float16)0.f;
}

// ---------------------------------------------------------------------------
// C[M,N] = act( A[M,K](f16) x W[N,K](f16)^T + bias[N] ), f16 output.
// Block: 256 threads = 8 waves (2 M x 4 N); block tile 128x128; wave 64x32.
// K staged through LDS in 32-wide chunks, double buffered.
// LDS rows padded to 40 halfwords (20 dwords) -> conflict-free fragment reads.
// ---------------------------------------------------------------------------
#define LSTR 40

__global__ __launch_bounds__(256) void gemm_wmma(
    const _Float16* __restrict__ A, const _Float16* __restrict__ W,
    const float* __restrict__ bias, _Float16* __restrict__ C,
    int M, int N, int K, int relu)
{
  __shared__ __align__(16) _Float16 sA[2][128][LSTR];
  __shared__ __align__(16) _Float16 sB[2][128][LSTR];

  const int tid    = threadIdx.x;
  const int lane   = tid & 31;
  const int wave   = tid >> 5;
  const int l15    = lane & 15;
  const int khalf  = lane >> 4;
  const int m_blk  = blockIdx.y * 128;
  const int n_blk  = blockIdx.x * 128;
  const int m_wave = (wave >> 2) * 64;
  const int n_wave = (wave & 3) * 32;

  // Staging: 256 threads move 128 rows x 32 halfwords (4 x 16B chunks per row)
  // for each of A and B: 2 chunks of each per thread.
  const int srow = tid >> 2;        // 0..63
  const int ssub = (tid & 3) * 8;   // halfword offset in row: 0,8,16,24

  v8f acc[4][2] = {{{}, {}}, {{}, {}}, {{}, {}}, {{}, {}}};

  auto stage = [&](int buf, int k0) {
    const _Float16* ga = A + (size_t)(m_blk + srow) * K + k0 + ssub;
    const _Float16* gb = W + (size_t)(n_blk + srow) * K + k0 + ssub;
    copy16_g2l(ga,                  &sA[buf][srow][ssub]);
    copy16_g2l(ga + (size_t)64 * K, &sA[buf][srow + 64][ssub]);
    copy16_g2l(gb,                  &sB[buf][srow][ssub]);
    copy16_g2l(gb + (size_t)64 * K, &sB[buf][srow + 64][ssub]);
  };

  auto compute = [&](int buf) {
    v16h a[4], b[2];
    #pragma unroll
    for (int i = 0; i < 4; ++i) {
      const int r = m_wave + i * 16 + l15;
      a[i] = frag_from(&sA[buf][r][khalf * 8], &sA[buf][r][16 + khalf * 8]);
    }
    #pragma unroll
    for (int j = 0; j < 2; ++j) {
      const int r = n_wave + j * 16 + l15;
      b[j] = frag_from(&sB[buf][r][khalf * 8], &sB[buf][r][16 + khalf * 8]);
    }
    #pragma unroll
    for (int i = 0; i < 4; ++i)
      #pragma unroll
      for (int j = 0; j < 2; ++j)
        acc[i][j] = wmma16x16x32(a[i], b[j], acc[i][j]);
  };

  const int nk = K >> 5;
  stage(0, 0);
  wait_async_copies();
  __syncthreads();

  int buf = 0;
  for (int kc = 0; kc < nk; ++kc) {
    if (kc + 1 < nk) stage(buf ^ 1, (kc + 1) << 5);  // prefetch next chunk
    compute(buf);                                     // overlaps with copies
    wait_async_copies();
    __syncthreads();   // next buffer ready; everyone done reading this one
    buf ^= 1;
  }

  // C/D layout (32-bit 16x16): lane n = lane&15, element r -> m = r + 8*(lane>>4)
  #pragma unroll
  for (int i = 0; i < 4; ++i) {
    #pragma unroll
    for (int j = 0; j < 2; ++j) {
      const int n  = n_blk + n_wave + j * 16 + l15;
      const int m0 = m_blk + m_wave + i * 16 + 8 * khalf;
      const float bv = bias[n];
      #pragma unroll
      for (int r = 0; r < 8; ++r) {
        float v = acc[i][j][r] + bv;
        if (relu) v = v > 0.f ? v : 0.f;
        C[(size_t)(m0 + r) * N + n] = (_Float16)v;
      }
    }
  }
}

// ---------------------------------------------------------------------------
// Fused embedding gather + pairwise interaction.
// One wave per batch row: T (27x128, padded 32x128 f16) in LDS, Z = T T^T via
// 16 WMMAs, pack R[b] = [x(128), tril(Z,-1)(351), 0(pad)] -> 480 f16.
// ---------------------------------------------------------------------------
#define TSTR 136  // padded f16 row stride (272B) to avoid bank conflicts

__global__ __launch_bounds__(128) void interact_wmma(
    const _Float16* __restrict__ X, const float* __restrict__ emb,
    const int* __restrict__ lS_i, _Float16* __restrict__ R)
{
  __shared__ __align__(16) _Float16 Tl[4][32][TSTR];
  __shared__ float Zl[4][32][33];

  const int lane = threadIdx.x & 31;
  const int wave = threadIdx.x >> 5;
  const int b    = blockIdx.x * 4 + wave;

  _Float16 (*T)[TSTR] = Tl[wave];
  float    (*Z)[33]   = Zl[wave];

  // Row 0: bottom-MLP output x[b]
  {
    const _Float16* xr = X + (size_t)b * 128;
    #pragma unroll
    for (int e = 0; e < 4; ++e) T[0][lane * 4 + e] = xr[lane * 4 + e];
  }
  // Rows 1..26: gathered embedding rows (f32 -> f16)
  for (int t = 0; t < 26; ++t) {
    const int idx = lS_i[t * 16384 + b];
    const float4 v = *(const float4*)(emb + ((size_t)t * 100000 + (size_t)idx) * 128
                                      + lane * 4);
    T[1 + t][lane * 4 + 0] = (_Float16)v.x;
    T[1 + t][lane * 4 + 1] = (_Float16)v.y;
    T[1 + t][lane * 4 + 2] = (_Float16)v.z;
    T[1 + t][lane * 4 + 3] = (_Float16)v.w;
  }
  #pragma unroll
  for (int rr = 27; rr < 32; ++rr)
    #pragma unroll
    for (int e = 0; e < 4; ++e) T[rr][lane * 4 + e] = (_Float16)0.f;
  __syncthreads();

  const int khalf = lane >> 4;
  const int l15   = lane & 15;
  v8f acc00 = {}, acc01 = {}, acc10 = {}, acc11 = {};

  #pragma unroll
  for (int k0 = 0; k0 < 128; k0 += 32) {
    v16h a0 = frag_from(&T[l15][k0 + khalf * 8],      &T[l15][k0 + 16 + khalf * 8]);
    v16h a1 = frag_from(&T[l15 + 16][k0 + khalf * 8], &T[l15 + 16][k0 + 16 + khalf * 8]);
    v16h b0 = frag_from(&T[l15][k0 + khalf * 8],      &T[l15][k0 + 16 + khalf * 8]);
    v16h b1 = frag_from(&T[l15 + 16][k0 + khalf * 8], &T[l15 + 16][k0 + 16 + khalf * 8]);
    acc00 = wmma16x16x32(a0, b0, acc00);
    acc01 = wmma16x16x32(a0, b1, acc01);
    acc10 = wmma16x16x32(a1, b0, acc10);
    acc11 = wmma16x16x32(a1, b1, acc11);
  }

  #pragma unroll
  for (int r = 0; r < 8; ++r) {
    Z[r + 8 * khalf +  0][l15 +  0] = acc00[r];
    Z[r + 8 * khalf +  0][l15 + 16] = acc01[r];
    Z[r + 8 * khalf + 16][l15 +  0] = acc10[r];
    Z[r + 8 * khalf + 16][l15 + 16] = acc11[r];
  }
  __syncthreads();

  _Float16* Rr = R + (size_t)b * 480;
  for (int e = lane; e < 128; e += 32) Rr[e] = T[0][e];
  // 351 strict-lower-tri entries + 1 zero pad = 352 = 11*32
  for (int p = lane; p < 352; p += 32) {
    if (p == 351) { Rr[479] = (_Float16)0.f; continue; }
    int i = 1;
    while (((i * (i + 1)) >> 1) <= p) ++i;         // tril_indices(27, k=-1) order
    const int j = p - ((i * (i - 1)) >> 1);
    Rr[128 + p] = (_Float16)Z[i][j];
  }
}

// ---------------------------------------------------------------------------
// Final 256 -> 1 layer + sigmoid (f32 VALU, trivial).
// ---------------------------------------------------------------------------
__global__ __launch_bounds__(256) void final_dot(
    const _Float16* __restrict__ A, const float* __restrict__ w,
    const float* __restrict__ bias, float* __restrict__ out)
{
  const int b = blockIdx.x * 256 + threadIdx.x;
  const _Float16* row = A + (size_t)b * 256;
  float s = bias[0];
  #pragma unroll 8
  for (int k = 0; k < 256; ++k) s += (float)row[k] * w[k];
  out[b] = 1.f / (1.f + __expf(-s));
}

// ---------------------------------------------------------------------------
// Host orchestration
// ---------------------------------------------------------------------------
extern "C" void kernel_launch(void* const* d_in, const int* in_sizes, int n_in,
                              void* d_out, int out_size, void* d_ws, size_t ws_size,
                              hipStream_t stream) {
  (void)in_sizes; (void)n_in; (void)out_size; (void)ws_size;
  constexpr int B = 16384;

  const float* dense_x = (const float*)d_in[0];
  /* d_in[1] = lS_o (arange, unused) */
  const int*   lS_i    = (const int*)  d_in[2];
  const float* emb     = (const float*)d_in[3];
  const float* bW0 = (const float*)d_in[4];   // 512 x 13
  const float* bW1 = (const float*)d_in[5];   // 256 x 512
  const float* bW2 = (const float*)d_in[6];   // 128 x 256
  const float* bb0 = (const float*)d_in[7];
  const float* bb1 = (const float*)d_in[8];
  const float* bb2 = (const float*)d_in[9];
  const float* tW0 = (const float*)d_in[10];  // 1024 x 479
  const float* tW1 = (const float*)d_in[11];  // 1024 x 1024
  const float* tW2 = (const float*)d_in[12];  // 512 x 1024
  const float* tW3 = (const float*)d_in[13];  // 256 x 512
  const float* tW4 = (const float*)d_in[14];  // 1 x 256
  const float* tb0 = (const float*)d_in[15];
  const float* tb1 = (const float*)d_in[16];
  const float* tb2 = (const float*)d_in[17];
  const float* tb3 = (const float*)d_in[18];
  const float* tb4 = (const float*)d_in[19];

  char* ws = (char*)d_ws;
  size_t off = 0;
  auto alloc = [&](size_t bytes) -> size_t {
    size_t o = off;
    off += (bytes + 255) & ~(size_t)255;
    return o;
  };
  const size_t oWd0 = alloc((size_t)512 * 32 * 2);
  const size_t oWd1 = alloc((size_t)256 * 512 * 2);
  const size_t oWd2 = alloc((size_t)128 * 256 * 2);
  const size_t oWt0 = alloc((size_t)1024 * 480 * 2);
  const size_t oWt1 = alloc((size_t)1024 * 1024 * 2);
  const size_t oWt2 = alloc((size_t)512 * 1024 * 2);
  const size_t oWt3 = alloc((size_t)256 * 512 * 2);
  const size_t oP   = alloc((size_t)B * 1024 * 2);  // ping
  const size_t oQ   = alloc((size_t)B * 1024 * 2);  // pong

  _Float16* Wd0 = (_Float16*)(ws + oWd0);
  _Float16* Wd1 = (_Float16*)(ws + oWd1);
  _Float16* Wd2 = (_Float16*)(ws + oWd2);
  _Float16* Wt0 = (_Float16*)(ws + oWt0);
  _Float16* Wt1 = (_Float16*)(ws + oWt1);
  _Float16* Wt2 = (_Float16*)(ws + oWt2);
  _Float16* Wt3 = (_Float16*)(ws + oWt3);
  _Float16* P   = (_Float16*)(ws + oP);
  _Float16* Q   = (_Float16*)(ws + oQ);

  auto cvt = [&](const float* s, _Float16* d, int rows, int kin, int kp) {
    const int n = rows * kp;
    cvt_pad<<<dim3((n + 255) / 256), dim3(256), 0, stream>>>(s, d, rows, kin, kp);
  };
  auto gemm = [&](const _Float16* A, const _Float16* W, const float* bias,
                  _Float16* C, int N, int K, int relu) {
    gemm_wmma<<<dim3(N / 128, B / 128), dim3(256), 0, stream>>>(
        A, W, bias, C, B, N, K, relu);
  };

  // Weight / input conversion (f32 -> f16, K padded to multiple of 32)
  cvt(dense_x, Q,   B,    13,   32);    // Xd lives in Q
  cvt(bW0,     Wd0, 512,  13,   32);
  cvt(bW1,     Wd1, 256,  512,  512);
  cvt(bW2,     Wd2, 128,  256,  256);
  cvt(tW0,     Wt0, 1024, 479,  480);
  cvt(tW1,     Wt1, 1024, 1024, 1024);
  cvt(tW2,     Wt2, 512,  1024, 1024);
  cvt(tW3,     Wt3, 256,  512,  512);

  // Bottom MLP: 13(->32) -> 512 -> 256 -> 128, ReLU
  gemm(Q, Wd0, bb0, P, 512, 32,  1);    // A1 in P
  gemm(P, Wd1, bb1, Q, 256, 512, 1);    // A2 in Q (Xd dead)
  gemm(Q, Wd2, bb2, P, 128, 256, 1);    // X  in P (A1 dead)

  // Gather + interactions -> R (B x 480 f16) in Q (A2 dead)
  interact_wmma<<<dim3(B / 4), dim3(128), 0, stream>>>(P, emb, lS_i, Q);

  // Top MLP: 479(->480) -> 1024 -> 1024 -> 512 -> 256 -> 1
  gemm(Q, Wt0, tb0, P, 1024, 480,  1);  // T1 in P (X dead)
  gemm(P, Wt1, tb1, Q, 1024, 1024, 1);  // T2 in Q (R dead)
  gemm(Q, Wt2, tb2, P, 512,  1024, 1);  // T3 in P
  gemm(P, Wt3, tb3, Q, 256,  512,  1);  // T4 in Q
  final_dot<<<dim3(B / 256), dim3(256), 0, stream>>>(Q, tW4, tb4, (float*)d_out);
}
